// ManualBLSTM_1288490189443
// MI455X (gfx1250) — matrix operand order repaired and checked by reading, not compiled, not run on hardware
//
#include <hip/hip_runtime.h>
#include <hip/hip_bf16.h>

typedef __attribute__((ext_vector_type(16))) __bf16 v16bf;
typedef __attribute__((ext_vector_type(8)))  float  v8f;

#define BB 8
#define SS 256
#define HH 512

__device__ __forceinline__ v16bf pack_bf16(uint4 lo, uint4 hi) {
    union { v16bf v; uint4 u[2]; } t;
    t.u[0] = lo; t.u[1] = hi;
    return t.v;
}

// CDNA5 async global->LDS copy (tracked by ASYNCcnt). VDST = LDS byte address
// (low 32 bits of the generic shared pointer), VADDR = 64-bit global address.
__device__ __forceinline__ void async_load_b128(uint32_t lds_off, const void* gaddr) {
    asm volatile("global_load_async_to_lds_b128 %0, %1, off"
                 :: "v"(lds_off), "v"(gaddr)
                 : "memory");
}
__device__ __forceinline__ void wait_async_le4() {
    asm volatile("s_wait_asynccnt 0x4" ::: "memory");
}
__device__ __forceinline__ void wait_async_0() {
    asm volatile("s_wait_asynccnt 0x0" ::: "memory");
}

// ---------------------------------------------------------------------------
// Embedding gather: out[row][0:512] = emb[ids[row]][0:512], row = b*S+s
// ---------------------------------------------------------------------------
__global__ __launch_bounds__(128)
void embed_kernel(const int* __restrict__ ids, const float* __restrict__ emb,
                  float* __restrict__ out) {
    const int row = blockIdx.x;
    const int id  = ids[row];
    const float4* src = (const float4*)(emb + (size_t)id  * 512);
    float4*       dst = (float4*)      (out + (size_t)row * 512);
    dst[threadIdx.x] = src[threadIdx.x];   // 128 threads * 4 floats = 512
}

// ---------------------------------------------------------------------------
// fp32 -> bf16 elementwise convert
// ---------------------------------------------------------------------------
__global__ __launch_bounds__(256)
void f32_to_bf16_kernel(const float* __restrict__ in, __bf16* __restrict__ out,
                        size_t n) {
    size_t i = (size_t)blockIdx.x * blockDim.x + threadIdx.x;
    size_t stride = (size_t)gridDim.x * blockDim.x;
    for (; i < n; i += stride) out[i] = (__bf16)in[i];
}

// ---------------------------------------------------------------------------
// bf16 GEMM: C[M,N] = A[M,K] * B[N,K]^T + bias[N], fp32 accumulate via WMMA.
// Block tile 128x128, BK=32, 256 threads (8 waves as 2x4), wave tile 64x32.
// Double-buffered LDS fed by global_load_async_to_lds_b128 (ASYNCcnt).
// ---------------------------------------------------------------------------
#define BM 128
#define BN 128
#define BK 32
#define LDP (BK + 8)   // LDS row stride (bf16 elems), 80B = 16B aligned

__global__ __launch_bounds__(256)
void gemm_bf16_kernel(const __bf16* __restrict__ A, const __bf16* __restrict__ B,
                      const float* __restrict__ bias, float* __restrict__ C,
                      int M, int N, int K) {
    __shared__ __align__(16) __bf16 As[2][BM][LDP];
    __shared__ __align__(16) __bf16 Bs[2][BN][LDP];

    const int tid  = threadIdx.x;
    const int lane = tid & 31;
    const int wid  = tid >> 5;
    const int wave_m = wid >> 2;   // 0..1  (64 rows each)
    const int wave_n = wid & 3;    // 0..3  (32 cols each)
    const int m0 = blockIdx.y * BM;
    const int n0 = blockIdx.x * BN;

    // This thread's two staging chunks (16B each, per matrix)
    const int chunk0 = tid;              // 0..255
    const int chunk1 = tid + 256;        // 256..511
    const int row0   = chunk0 >> 2, col0 = (chunk0 & 3) << 3;
    const int row1   = chunk1 >> 2, col1 = (chunk1 & 3) << 3;

    auto issue_tile = [&](int buf, int kb) {
        async_load_b128((uint32_t)(size_t)&As[buf][row0][col0],
                        A + (size_t)(m0 + row0) * K + kb + col0);
        async_load_b128((uint32_t)(size_t)&Bs[buf][row0][col0],
                        B + (size_t)(n0 + row0) * K + kb + col0);
        async_load_b128((uint32_t)(size_t)&As[buf][row1][col1],
                        A + (size_t)(m0 + row1) * K + kb + col1);
        async_load_b128((uint32_t)(size_t)&Bs[buf][row1][col1],
                        B + (size_t)(n0 + row1) * K + kb + col1);
    };

    const v8f vzero = {};
    v8f acc[4][2];
#pragma unroll
    for (int mi = 0; mi < 4; ++mi)
#pragma unroll
        for (int ni = 0; ni < 2; ++ni) acc[mi][ni] = vzero;

    const int a_c0 = (lane < 16) ? 0 : 8;   // A frag K split {0..7,16..23}/{8..15,24..31}
    const int b_c0 = (lane < 16) ? 0 : 8;   // B frag K split {0..7,16..23}/{8..15,24..31}

    issue_tile(0, 0);
    int buf = 0;
    for (int kb = 0; kb < K; kb += BK, buf ^= 1) {
        const int nkb = kb + BK;
        if (nkb < K) {
            issue_tile(buf ^ 1, nkb);   // overlap next tile's DMA with compute
            wait_async_le4();           // oldest 4 (current tile) complete in-order
        } else {
            wait_async_0();
        }
        __syncthreads();

        v16bf af[4];
#pragma unroll
        for (int mi = 0; mi < 4; ++mi) {
            const int r = wave_m * 64 + mi * 16 + (lane & 15);
            uint4 lo = *(const uint4*)&As[buf][r][a_c0];
            uint4 hi = *(const uint4*)&As[buf][r][a_c0 + 16];
            af[mi] = pack_bf16(lo, hi);
        }
        v16bf bf[2];
#pragma unroll
        for (int ni = 0; ni < 2; ++ni) {
            const int r = wave_n * 32 + ni * 16 + (lane & 15);
            uint4 lo = *(const uint4*)&Bs[buf][r][b_c0];
            uint4 hi = *(const uint4*)&Bs[buf][r][b_c0 + 16];
            bf[ni] = pack_bf16(lo, hi);
        }
#pragma unroll
        for (int mi = 0; mi < 4; ++mi)
#pragma unroll
            for (int ni = 0; ni < 2; ++ni)
                acc[mi][ni] = __builtin_amdgcn_wmma_f32_16x16x32_bf16(
                    false, af[mi], false, bf[ni], (short)0, acc[mi][ni], false, false);
        __syncthreads();   // protect buf from next iteration's async writes
    }

    // ---- epilogue: VGPR r holds M=r (lanes 0-15) / M=8+r (lanes 16-31)
#pragma unroll
    for (int mi = 0; mi < 4; ++mi)
#pragma unroll
        for (int ni = 0; ni < 2; ++ni) {
            const int mbase = m0 + wave_m * 64 + mi * 16 + ((lane >> 4) << 3);
            const int n     = n0 + wave_n * 32 + ni * 16 + (lane & 15);
            const float bv  = bias ? bias[n] : 0.0f;
#pragma unroll
            for (int r = 0; r < 8; ++r)
                C[(size_t)(mbase + r) * N + n] = acc[mi][ni][r] + bv;
        }
}

// ---------------------------------------------------------------------------
// LSTM scan: one workgroup per direction (grid=2). 512 threads = 16 waves.
// Per step: gates[8,2048] = h(16x512,bf16,LDS) @ U^T(2048x512,bf16,L2) via
// WMMA (each wave: 8 N-tiles x 16 K-steps), then fused elementwise update.
// Cell state c lives in registers (8 values/thread).
// ---------------------------------------------------------------------------
#define HPAD 520   // bf16 row stride for h in LDS (1040B, 16B aligned)
#define SCAN_LDS_BYTES (8 * 2048 * 4 + 16 * HPAD * 2)

__global__ __launch_bounds__(512)
void lstm_scan_kernel(const float* __restrict__ xg_f, const float* __restrict__ xg_b,
                      const __bf16* __restrict__ U_f, const __bf16* __restrict__ U_b,
                      float* __restrict__ out) {
    extern __shared__ char smem[];
    float*  gates = (float*)smem;                      // [8][2048]
    __bf16* hbf   = (__bf16*)(smem + 8 * 2048 * 4);    // [16][HPAD], rows 8..15 zero

    const int dir = blockIdx.x;
    const float*  xg = dir ? xg_b : xg_f;
    const __bf16* U  = dir ? U_b  : U_f;
    const int off    = dir ? HH : 0;

    const int tid  = threadIdx.x;
    const int lane = tid & 31;
    const int wid  = tid >> 5;

    for (int i = tid; i < 16 * HPAD; i += 512) hbf[i] = (__bf16)0.0f;
    float c[8];
#pragma unroll
    for (int k = 0; k < 8; ++k) c[k] = 0.0f;
    __syncthreads();

    const int a_c0 = (lane < 16) ? 0 : 8;
    const int rowa = (lane & 15) * HPAD;
    const v8f vzero = {};

    for (int tt = 0; tt < SS; ++tt) {
        const int time = dir ? (SS - 1 - tt) : tt;

        // ---- recurrent GEMM: each wave owns N-tiles wid*8 .. wid*8+7
        v8f acc[8];
#pragma unroll
        for (int j = 0; j < 8; ++j) acc[j] = vzero;

        for (int kb = 0; kb < HH; kb += 32) {
            uint4 alo = *(const uint4*)&hbf[rowa + kb + a_c0];
            uint4 ahi = *(const uint4*)&hbf[rowa + kb + a_c0 + 16];
            const v16bf af = pack_bf16(alo, ahi);
            const int bk0 = kb + ((lane < 16) ? 0 : 8);
#pragma unroll
            for (int j = 0; j < 8; ++j) {
                const int n = (wid * 8 + j) * 16 + (lane & 15);
                const __bf16* bp = U + (size_t)n * HH + bk0;
                uint4 blo = *(const uint4*)bp;
                uint4 bhi = *(const uint4*)(bp + 16);
                const v16bf bfr = pack_bf16(blo, bhi);
                acc[j] = __builtin_amdgcn_wmma_f32_16x16x32_bf16(
                    false, af, false, bfr, (short)0, acc[j], false, false);
            }
        }
        // lanes 0..15 hold batch rows 0..7 (VGPR r <-> M=r)
        if (lane < 16) {
#pragma unroll
            for (int j = 0; j < 8; ++j) {
                const int ncol = (wid * 8 + j) * 16 + lane;
#pragma unroll
                for (int r = 0; r < 8; ++r)
                    gates[r * 2048 + ncol] = acc[j][r];
            }
        }
        __syncthreads();

        // ---- fused elementwise LSTM cell update (gate order i,f,o,g)
#pragma unroll
        for (int k = 0; k < 8; ++k) {
            const int idx = tid + k * 512;       // 0..4095
            const int b = idx >> 9;              // 0..7
            const int j = idx & 511;             // 0..511
            const float* xr = xg + (size_t)(b * SS + time) * 2048;
            float iv = gates[b * 2048 + j]         + xr[j];
            float fv = gates[b * 2048 + 512 + j]   + xr[512 + j];
            float ov = gates[b * 2048 + 1024 + j]  + xr[1024 + j];
            float gv = gates[b * 2048 + 1536 + j]  + xr[1536 + j];
            iv = 1.0f / (1.0f + __expf(-iv));
            fv = 1.0f / (1.0f + __expf(-fv));
            ov = 1.0f / (1.0f + __expf(-ov));
            gv = tanhf(gv);
            c[k] = fv * c[k] + iv * gv;
            const float h = ov * tanhf(c[k]);
            hbf[b * HPAD + j] = (__bf16)h;
            out[(size_t)(b * SS + time) * (2 * HH) + off + j] = h;
        }
        __syncthreads();
    }
}

// ---------------------------------------------------------------------------
// Host side
// ---------------------------------------------------------------------------
static void convert(const float* in, __bf16* out, size_t n, hipStream_t s) {
    int blocks = (int)((n + 2047) / 2048);
    if (blocks > 4096) blocks = 4096;
    f32_to_bf16_kernel<<<blocks, 256, 0, s>>>(in, out, n);
}

extern "C" void kernel_launch(void* const* d_in, const int* in_sizes, int n_in,
                              void* d_out, int out_size, void* d_ws, size_t ws_size,
                              hipStream_t stream) {
    (void)in_sizes; (void)n_in; (void)out_size; (void)ws_size;

    const int*   ids    = (const int*)d_in[0];
    const float* emb    = (const float*)d_in[1];
    const float* fw0_W  = (const float*)d_in[2];
    const float* fw0_U  = (const float*)d_in[3];
    const float* fw0_b  = (const float*)d_in[4];
    const float* bw0_W  = (const float*)d_in[5];
    const float* bw0_U  = (const float*)d_in[6];
    const float* bw0_b  = (const float*)d_in[7];
    const float* fw1_W  = (const float*)d_in[8];
    const float* fw1_U  = (const float*)d_in[9];
    const float* fw1_b  = (const float*)d_in[10];
    const float* bw1_W  = (const float*)d_in[11];
    const float* bw1_U  = (const float*)d_in[12];
    const float* bw1_b  = (const float*)d_in[13];
    const float* w_out  = (const float*)d_in[14];
    const float* b_out  = (const float*)d_in[15];

    const int M = BB * SS;        // 2048 token rows
    char* ws = (char*)d_ws;
    size_t off = 0;
    auto alloc = [&](size_t bytes) -> void* {
        off = (off + 255) & ~(size_t)255;
        void* p = ws + off;
        off += bytes;
        return p;
    };

    float*  cur0   = (float*) alloc((size_t)M * 512 * 4);
    float*  curA   = (float*) alloc((size_t)M * 1024 * 4);   // layer0 output
    float*  curB   = (float*) alloc((size_t)M * 1024 * 4);   // layer1 output
    float*  xgF    = (float*) alloc((size_t)M * 2048 * 4);
    float*  xgB    = (float*) alloc((size_t)M * 2048 * 4);
    __bf16* bfA    = (__bf16*)alloc((size_t)M * 1024 * 2);   // activations bf16
    __bf16* bfW    = (__bf16*)alloc((size_t)2048 * 1024 * 2);
    __bf16* bfUf   = (__bf16*)alloc((size_t)2048 * 512 * 2);
    __bf16* bfUb   = (__bf16*)alloc((size_t)2048 * 512 * 2);
    __bf16* bfWout = (__bf16*)alloc((size_t)32000 * 1024 * 2);

    hipFuncSetAttribute(reinterpret_cast<const void*>(lstm_scan_kernel),
                        hipFuncAttributeMaxDynamicSharedMemorySize, SCAN_LDS_BYTES);

    // 1. embedding
    embed_kernel<<<M, 128, 0, stream>>>(ids, emb, cur0);
    convert(cur0, bfA, (size_t)M * 512, stream);

    // 2. layer 0: xg precompute (K=512), then scan
    convert(fw0_W, bfW, (size_t)2048 * 512, stream);
    gemm_bf16_kernel<<<dim3(2048 / BN, M / BM), 256, 0, stream>>>(
        bfA, bfW, fw0_b, xgF, M, 2048, 512);
    convert(bw0_W, bfW, (size_t)2048 * 512, stream);
    gemm_bf16_kernel<<<dim3(2048 / BN, M / BM), 256, 0, stream>>>(
        bfA, bfW, bw0_b, xgB, M, 2048, 512);
    convert(fw0_U, bfUf, (size_t)2048 * 512, stream);
    convert(bw0_U, bfUb, (size_t)2048 * 512, stream);
    lstm_scan_kernel<<<2, 512, SCAN_LDS_BYTES, stream>>>(xgF, xgB, bfUf, bfUb, curA);

    // 3. layer 1: xg precompute (K=1024), then scan
    convert(curA, bfA, (size_t)M * 1024, stream);
    convert(fw1_W, bfW, (size_t)2048 * 1024, stream);
    gemm_bf16_kernel<<<dim3(2048 / BN, M / BM), 256, 0, stream>>>(
        bfA, bfW, fw1_b, xgF, M, 2048, 1024);
    convert(bw1_W, bfW, (size_t)2048 * 1024, stream);
    gemm_bf16_kernel<<<dim3(2048 / BN, M / BM), 256, 0, stream>>>(
        bfA, bfW, bw1_b, xgB, M, 2048, 1024);
    convert(fw1_U, bfUf, (size_t)2048 * 512, stream);
    convert(bw1_U, bfUb, (size_t)2048 * 512, stream);
    lstm_scan_kernel<<<2, 512, SCAN_LDS_BYTES, stream>>>(xgF, xgB, bfUf, bfUb, curB);

    // 4. LM head: logits[M,32000] = curB * w_out^T + b_out  (dominant GEMM)
    convert(curB, bfA, (size_t)M * 1024, stream);
    convert(w_out, bfWout, (size_t)32000 * 1024, stream);
    gemm_bf16_kernel<<<dim3(32000 / BN, M / BM), 256, 0, stream>>>(
        bfA, bfWout, b_out, (float*)d_out, M, 32000, 1024);

    // 5. append cur (layer-1 output) after logits in d_out
    hipMemcpyAsync((float*)d_out + (size_t)M * 32000, curB,
                   (size_t)M * 1024 * 4, hipMemcpyDeviceToDevice, stream);
}
